// cnnblstm_33045478375963
// MI455X (gfx1250) — matrix operand order repaired and checked by reading, not compiled, not run on hardware
//
#include <hip/hip_runtime.h>
#include <hip/hip_bf16.h>

// ---------------- problem constants ----------------
#define BATCH   256
#define T_IN    800
#define F_INC   3
#define C_CONV  128
#define KCONV   15
#define TT      393     // pooled time steps
#define HH      150     // lstm hidden
#define HP      160     // padded hidden (k dim, mult of 32)
#define G4      600     // 4*H gate columns
#define GP      640     // padded gate columns: 40 tiles of 16 -> 5 per wave, no guards
#define NT      40      // n tiles
#define NTW     5       // n-tiles per wave (40 / 8 waves)
#define KT1     4       // k tiles for input proj (C_CONV=128)
#define KT2     5       // k tiles for recurrent (HP=160)
#define MROWS   32      // batch rows per block
#define NIT     19      // ceil(32*150/256)

typedef __attribute__((ext_vector_type(16))) __bf16 v16bf;
typedef __attribute__((ext_vector_type(8)))  float  v8f;

union AF { unsigned int u[8]; v16bf v; };

__device__ __forceinline__ unsigned short f2bf(float f) {
    unsigned int u = __builtin_bit_cast(unsigned int, f);
    unsigned int r = ((u >> 16) & 1u) + 0x7fffu;   // round to nearest even
    return (unsigned short)((u + r) >> 16);
}

__device__ __forceinline__ float hsig(float x) {
    return fminf(fmaxf(0.2f * x + 0.5f, 0.0f), 1.0f);
}

// A-fragment (16x32 bf16, MxK): lane lo holds K {kb..kb+15},
// lane hi holds K {kb+16..kb+31}; 2 consecutive K per dword.
__device__ __forceinline__ v16bf load_a16(const unsigned short* rowbase, int kb, int laneHi) {
    AF f;
    const uint4* q = (const uint4*)(rowbase + kb + laneHi * 16);
    uint4 lo = q[0];
    uint4 hi = q[1];   // +8 bf16 = +16B
    f.u[0] = lo.x; f.u[1] = lo.y; f.u[2] = lo.z; f.u[3] = lo.w;
    f.u[4] = hi.x; f.u[5] = hi.y; f.u[6] = hi.z; f.u[7] = hi.w;
    return f.v;
}

// B-fragment from pre-swizzled weights: per (ktile, column g) 16 dwords;
// dwords 0..7 = lane-lo (K 0..15 of tile), 8..15 = lane-hi (K 16..31).
__device__ __forceinline__ v16bf load_b16(const unsigned int* base, int kt, int g, int laneHi) {
    AF f;
    const uint4* q = (const uint4*)(base + ((kt * GP + g) * 16 + laneHi * 8));
    uint4 a = q[0];
    uint4 b = q[1];
    f.u[0] = a.x; f.u[1] = a.y; f.u[2] = a.z; f.u[3] = a.w;
    f.u[4] = b.x; f.u[5] = b.y; f.u[6] = b.z; f.u[7] = b.w;
    return f.v;
}

// ---------------- kernel 1: weight swizzle to bf16 B-fragment layout ----------
// wihswz: [dir][KT1][GP][16 dwords]   (B = w_ih^T, K=128)
// whhswz: [dir][KT2][GP][16 dwords]   (B = w_hh^T, K=160, rows>=150 zero)
#define NIH (2 * KT1 * GP * 16)    // 81,920 dwords
#define NHH (2 * KT2 * GP * 16)    // 102,400 dwords

__global__ void prep_weights_kernel(const float* __restrict__ wih_f,
                                    const float* __restrict__ wih_b,
                                    const float* __restrict__ whh_f,
                                    const float* __restrict__ whh_b,
                                    unsigned int* __restrict__ wihswz,
                                    unsigned int* __restrict__ whhswz) {
    int idx = blockIdx.x * blockDim.x + threadIdx.x;
    if (idx < NIH) {
        int d  = idx / (KT1 * GP * 16);
        int r  = idx - d * (KT1 * GP * 16);
        int kt = r / (GP * 16);
        int r2 = r - kt * (GP * 16);
        int g  = r2 >> 4;
        int dd = r2 & 15;
        int k0 = kt * 32 + (dd >> 3) * 16 + (dd & 7) * 2;
        const float* W = d ? wih_b : wih_f;   // [600][128]
        float f0 = (g < G4 && k0     < C_CONV) ? W[g * C_CONV + k0]     : 0.0f;
        float f1 = (g < G4 && k0 + 1 < C_CONV) ? W[g * C_CONV + k0 + 1] : 0.0f;
        wihswz[idx] = (unsigned int)f2bf(f0) | ((unsigned int)f2bf(f1) << 16);
    } else if (idx < NIH + NHH) {
        int l  = idx - NIH;
        int d  = l / (KT2 * GP * 16);
        int r  = l - d * (KT2 * GP * 16);
        int kt = r / (GP * 16);
        int r2 = r - kt * (GP * 16);
        int g  = r2 >> 4;
        int dd = r2 & 15;
        int k0 = kt * 32 + (dd >> 3) * 16 + (dd & 7) * 2;
        const float* W = d ? whh_b : whh_f;   // [600][150]
        float f0 = (g < G4 && k0     < HH) ? W[g * HH + k0]     : 0.0f;
        float f1 = (g < G4 && k0 + 1 < HH) ? W[g * HH + k0 + 1] : 0.0f;
        whhswz[l] = (unsigned int)f2bf(f0) | ((unsigned int)f2bf(f1) << 16);
    }
}

// ---------------- kernel 2: conv1d + relu + maxpool2 -> y[t][b][c] bf16 -------
__global__ void conv_pool_kernel(const float* __restrict__ x,      // [B][3][800]
                                 const float* __restrict__ cw,     // [128][3][15]
                                 const float* __restrict__ cb,     // [128]
                                 unsigned short* __restrict__ y) { // [TT][B][128] bf16
    __shared__ float xs[F_INC * T_IN];
    int b = blockIdx.x;
    int c = threadIdx.x;
    for (int i = c; i < F_INC * T_IN; i += C_CONV) xs[i] = x[b * (F_INC * T_IN) + i];
    float w[F_INC * KCONV];
#pragma unroll
    for (int i = 0; i < F_INC * KCONV; ++i) w[i] = cw[c * (F_INC * KCONV) + i];
    float bias = cb[c];
    __syncthreads();
    for (int t = 0; t < TT; ++t) {
        int p = 2 * t;
        float a0 = bias, a1 = bias;
#pragma unroll
        for (int i = 0; i < F_INC; ++i) {
#pragma unroll
            for (int k = 0; k < KCONV; ++k) {
                float wv = w[i * KCONV + k];
                a0 = fmaf(xs[i * T_IN + p + k],     wv, a0);
                a1 = fmaf(xs[i * T_IN + p + 1 + k], wv, a1);
            }
        }
        float v = fmaxf(fmaxf(a0, a1), 0.0f);
        y[(t * BATCH + b) * C_CONV + c] = f2bf(v);
    }
}

// ---------------- kernel 3: persistent bidirectional LSTM --------------------
// grid = 16 blocks (8 batch slices x 2 dirs), 256 threads (8 wave32).
// LDS: whh swizzled (204800B) | gates f32 32x640 (81920B) | h bf16 32x160
// (10240B) | bias sums (2560B)  => 299520 B dynamic LDS (<= 320KB/WGP).
#define LDS_WHH_OFF   0
#define LDS_GATE_OFF  204800
#define LDS_H_OFF     (204800 + 81920)
#define LDS_BSUM_OFF  (204800 + 81920 + 10240)
#define LDS_TOTAL     (204800 + 81920 + 10240 + 2560)

__global__ void __launch_bounds__(256, 1)
blstm_kernel(const unsigned short* __restrict__ y,       // [TT][B][128] bf16
             const unsigned int*  __restrict__ wihswz,   // [2][KT1*GP*16]
             const unsigned int*  __restrict__ whhswz,   // [2][KT2*GP*16]
             const float* __restrict__ bih_f, const float* __restrict__ bhh_f,
             const float* __restrict__ bih_b, const float* __restrict__ bhh_b,
             float* __restrict__ pooled) {               // [B][300]
    extern __shared__ char smem[];
    unsigned int*   whhL  = (unsigned int*)(smem + LDS_WHH_OFF);
    float*          gates = (float*)(smem + LDS_GATE_OFF);
    unsigned short* hb    = (unsigned short*)(smem + LDS_H_OFF);
    float*          bsum  = (float*)(smem + LDS_BSUM_OFF);

    const int tid  = threadIdx.x;
    const int wid  = tid >> 5;
    const int lane = tid & 31;
    const int laneHi = lane >> 4;
    const int lrow   = lane & 15;
    const int dir   = blockIdx.x & 1;
    const int b0    = (blockIdx.x >> 1) * MROWS;

    const unsigned int* wih  = wihswz + (size_t)dir * (KT1 * GP * 16);
    const unsigned int* whhG = whhswz + (size_t)dir * (KT2 * GP * 16);
    const float* bih = dir ? bih_b : bih_f;
    const float* bhh = dir ? bhh_b : bhh_f;

    // stage recurrent weights into LDS (25600 x uint2 == 204800 B; 100 iters)
    {
        const uint2* src = (const uint2*)whhG;
        uint2* dst = (uint2*)whhL;
        for (int i = tid; i < (KT2 * GP * 16) / 2; i += 256) dst[i] = src[i];
    }
    // zero h buffer (incl. padded cols 150..159)
    {
        unsigned int* hz = (unsigned int*)hb;
        for (int i = tid; i < MROWS * HP / 2; i += 256) hz[i] = 0u;
    }
    // bias sums
    for (int g = tid; g < G4; g += 256) bsum[g] = bih[g] + bhh[g];
    __syncthreads();

    float creg[NIT];
    float mreg[NIT];
#pragma unroll
    for (int it = 0; it < NIT; ++it) { creg[it] = 0.0f; mreg[it] = -1.0e30f; }

    for (int t = 0; t < TT; ++t) {
        int tt = dir ? (TT - 1 - t) : t;

        v8f acc[NTW][2];
#pragma unroll
        for (int i = 0; i < NTW; ++i) {
            v8f z = {0.f, 0.f, 0.f, 0.f, 0.f, 0.f, 0.f, 0.f};
            acc[i][0] = z; acc[i][1] = z;
        }

        // GEMM1: input projection, A = y[tt] rows (global, L2 resident)
        const unsigned short* yr0 = y + ((size_t)tt * BATCH + b0 + lrow) * C_CONV;
        const unsigned short* yr1 = yr0 + 16 * C_CONV;
#pragma unroll
        for (int kt = 0; kt < KT1; ++kt) {
            v16bf a0 = load_a16(yr0, kt * 32, laneHi);
            v16bf a1 = load_a16(yr1, kt * 32, laneHi);
#pragma unroll
            for (int i = 0; i < NTW; ++i) {
                int nt = wid + i * 8;                       // wave-uniform, always < NT
                v16bf bf = load_b16(wih, kt, nt * 16 + lrow, laneHi);
                acc[i][0] = __builtin_amdgcn_wmma_f32_16x16x32_bf16(
                    false, a0, false, bf, (short)0, acc[i][0], false, false);
                acc[i][1] = __builtin_amdgcn_wmma_f32_16x16x32_bf16(
                    false, a1, false, bf, (short)0, acc[i][1], false, false);
            }
        }
        // GEMM2: recurrence, A = h (LDS), B = whh (LDS)
#pragma unroll
        for (int kt = 0; kt < KT2; ++kt) {
            v16bf a0 = load_a16(hb + lrow * HP,        kt * 32, laneHi);
            v16bf a1 = load_a16(hb + (16 + lrow) * HP, kt * 32, laneHi);
#pragma unroll
            for (int i = 0; i < NTW; ++i) {
                int nt = wid + i * 8;
                v16bf bf = load_b16(whhL, kt, nt * 16 + lrow, laneHi);
                acc[i][0] = __builtin_amdgcn_wmma_f32_16x16x32_bf16(
                    false, a0, false, bf, (short)0, acc[i][0], false, false);
                acc[i][1] = __builtin_amdgcn_wmma_f32_16x16x32_bf16(
                    false, a1, false, bf, (short)0, acc[i][1], false, false);
            }
        }

        // scatter C tiles to LDS gates (C/D layout: vgpr r -> M = 8*laneHi + r)
#pragma unroll
        for (int i = 0; i < NTW; ++i) {
            int col = (wid + i * 8) * 16 + lrow;
#pragma unroll
            for (int mt = 0; mt < 2; ++mt) {
                int rbase = mt * 16 + laneHi * 8;
#pragma unroll
                for (int r = 0; r < 8; ++r)
                    gates[(rbase + r) * GP + col] = acc[i][mt][r];
            }
        }
        __syncthreads();

        // pointwise LSTM cell update (gate order i,f,g,o)
#pragma unroll
        for (int it = 0; it < NIT; ++it) {
            int e = tid + it * 256;
            if (e < MROWS * HH) {
                int row = e / HH;
                int j   = e - row * HH;
                float gi = gates[row * GP + j]            + bsum[j];
                float gf = gates[row * GP + HH + j]       + bsum[HH + j];
                float gg = gates[row * GP + 2 * HH + j]   + bsum[2 * HH + j];
                float go = gates[row * GP + 3 * HH + j]   + bsum[3 * HH + j];
                float ii = hsig(gi), ff = hsig(gf), oo = hsig(go);
                float c  = ff * creg[it] + ii * tanhf(gg);
                creg[it] = c;
                float h  = oo * tanhf(c);
                mreg[it] = fmaxf(mreg[it], h);
                hb[row * HP + j] = f2bf(h);
            }
        }
        __syncthreads();
    }

    // temporal max pooled output: [B][2H], forward = cols 0..149, back = 150..299
#pragma unroll
    for (int it = 0; it < NIT; ++it) {
        int e = tid + it * 256;
        if (e < MROWS * HH) {
            int row = e / HH;
            int j   = e - row * HH;
            pooled[(size_t)(b0 + row) * (2 * HH) + dir * HH + j] = mreg[it];
        }
    }
}

// ---------------- kernel 4: MLP head + softmax -------------------------------
__global__ void head_kernel(const float* __restrict__ pooled,  // [B][300]
                            const float* __restrict__ w1, const float* __restrict__ b1,
                            const float* __restrict__ w2, const float* __restrict__ b2,
                            float* __restrict__ out) {          // [B][10]
    __shared__ float pr[2 * HH];
    __shared__ float z[50];
    __shared__ float lg[10];
    __shared__ float red[2];
    int b = blockIdx.x, tid = threadIdx.x;
    for (int i = tid; i < 2 * HH; i += 64) pr[i] = pooled[(size_t)b * (2 * HH) + i];
    __syncthreads();
    for (int k = tid; k < 50; k += 64) {
        float s = b1[k];
        for (int i = 0; i < 2 * HH; ++i) s = fmaf(pr[i], w1[k * (2 * HH) + i], s);
        z[k] = fmaxf(s, 0.0f);
    }
    __syncthreads();
    if (tid < 10) {
        float s = b2[tid];
        for (int i = 0; i < 50; ++i) s = fmaf(z[i], w2[tid * 50 + i], s);
        lg[tid] = s;
    }
    __syncthreads();
    if (tid == 0) {
        float m = lg[0];
        for (int i = 1; i < 10; ++i) m = fmaxf(m, lg[i]);
        float s = 0.0f;
        for (int i = 0; i < 10; ++i) s += expf(lg[i] - m);
        red[0] = m; red[1] = s;
    }
    __syncthreads();
    if (tid < 10) out[b * 10 + tid] = expf(lg[tid] - red[0]) / red[1];
}

// ---------------- launcher ----------------------------------------------------
extern "C" void kernel_launch(void* const* d_in, const int* in_sizes, int n_in,
                              void* d_out, int out_size, void* d_ws, size_t ws_size,
                              hipStream_t stream) {
    const float* x      = (const float*)d_in[0];
    const float* conv_w = (const float*)d_in[1];
    const float* conv_b = (const float*)d_in[2];
    const float* wih_f  = (const float*)d_in[3];
    const float* whh_f  = (const float*)d_in[4];
    const float* bih_f  = (const float*)d_in[5];
    const float* bhh_f  = (const float*)d_in[6];
    const float* wih_b  = (const float*)d_in[7];
    const float* whh_b  = (const float*)d_in[8];
    const float* bih_b  = (const float*)d_in[9];
    const float* bhh_b  = (const float*)d_in[10];
    const float* w1     = (const float*)d_in[11];
    const float* b1     = (const float*)d_in[12];
    const float* w2     = (const float*)d_in[13];
    const float* b2     = (const float*)d_in[14];
    float* out = (float*)d_out;

    // workspace carve (all 256B aligned)
    const size_t Y_BYTES   = (size_t)TT * BATCH * C_CONV * 2;   // 25,755,648
    const size_t WIH_BYTES = (size_t)NIH * 4;                   //    327,680
    const size_t WHH_BYTES = (size_t)NHH * 4;                   //    409,600
    char* ws = (char*)d_ws;
    unsigned short* y       = (unsigned short*)(ws);
    unsigned int*   wihswz  = (unsigned int*)(ws + Y_BYTES);
    unsigned int*   whhswz  = (unsigned int*)(ws + Y_BYTES + WIH_BYTES);
    float*          pooled  = (float*)(ws + Y_BYTES + WIH_BYTES + WHH_BYTES);

    (void)hipFuncSetAttribute((const void*)blstm_kernel,
                              hipFuncAttributeMaxDynamicSharedMemorySize, LDS_TOTAL);

    int prep_threads = NIH + NHH;   // 184,320
    prep_weights_kernel<<<(prep_threads + 255) / 256, 256, 0, stream>>>(
        wih_f, wih_b, whh_f, whh_b, wihswz, whhswz);

    conv_pool_kernel<<<BATCH, C_CONV, 0, stream>>>(x, conv_w, conv_b, y);

    blstm_kernel<<<16, 256, LDS_TOTAL, stream>>>(
        y, wihswz, whhswz, bih_f, bhh_f, bih_b, bhh_b, pooled);

    head_kernel<<<BATCH, 64, 0, stream>>>(pooled, w1, b1, w2, b2, out);
}
